// Hierachical_Recurrent_Framework_45251775430857
// MI455X (gfx1250) — matrix operand order repaired and checked by reading, not compiled, not run on hardware
//
#include <hip/hip_runtime.h>
#include <cstddef>
#include <cstdint>

typedef __attribute__((ext_vector_type(16))) __bf16 v16bf;
typedef __attribute__((ext_vector_type(8)))  float  v8f;
typedef __attribute__((ext_vector_type(4)))  int    i32x4;
typedef __attribute__((ext_vector_type(2)))  int    i32x2;

#define B_SZ 2048
#define T_SZ 120
#define H_SZ 256
#define G3   768      // 3H
#define MT   16       // batch rows per block
#define NTPW 6        // 48 N-tiles / 8 waves
#define AST  776      // LDS A-row stride in halfs: 512 (max Din) + 256 (H) + 8 pad

// ---- explicit global-address-space accessors (force global_load/store) ----
typedef const __attribute__((address_space(1))) i32x4*  gcp_i4;
typedef const __attribute__((address_space(1))) i32x2*  gcp_i2;
typedef const __attribute__((address_space(1))) float*  gcp_f32;
typedef __attribute__((address_space(1))) __bf16*       gp_bf;
typedef __attribute__((address_space(1))) float*        gp_f32;

static __device__ __forceinline__ i32x4 ld_g128(const void* p) { return *(gcp_i4)p; }
static __device__ __forceinline__ i32x2 ld_g64 (const void* p) { return *(gcp_i2)p; }
static __device__ __forceinline__ float ld_gf  (const void* p) { return *(gcp_f32)p; }
static __device__ __forceinline__ void  st_gbf (void* p, __bf16 v) { *(gp_bf)p = v; }
static __device__ __forceinline__ void  st_gf  (void* p, float v)  { *(gp_f32)p = v; }

static __device__ __forceinline__ v8f wmma_bf16(v16bf a, v16bf b, v8f c) {
  return __builtin_amdgcn_wmma_f32_16x16x32_bf16(false, a, false, b, (short)0, c,
                                                 false, false);
}

// packed-weight fragment address: [tile][ks][lane][16 halfs], 32B/lane, 1KB/wave
static __device__ __forceinline__ v16bf load_bfrag(const __bf16* Wpk, int tile,
                                                   int KS, int ks, int lane) {
  const __bf16* pb = Wpk + ((((size_t)tile * KS + ks) * 32) + lane) * 16;
  v16bf b;
  ((i32x4*)&b)[0] = ld_g128(pb);
  ((i32x4*)&b)[1] = ld_g128(pb + 8);
  return b;
}

struct SMem {
  __bf16 A[MT][AST];   // staged [x_t | h_bf16] rows (WMMA A source)
  float  P[MT][1024];  // preacts: r(0..255) z(256..511) n_i(512..767) n_h(768..1023)
  float  H[MT][260];   // f32 hidden carry (recurrent bypass path precision)
};

// ---------------------------------------------------------------------------
// Persistent GRU scan over T=120 for a 16-row batch tile.
// Each wave owns 6 of 48 N-tiles of 3H=768; WMMA bf16 16x16x32, f32 accum.
// Weight fragments are re-read from L2 each step; pointer laundering below
// stops LICM from hoisting 120 steps' worth of loads into registers (which
// previously forced catastrophic scratch spills).
// ---------------------------------------------------------------------------
template <int D0, int D1, bool OUTF>
static __device__ void gru_scan_body(
    SMem& sm, int b0,
    const __bf16* __restrict__ x0, const __bf16* __restrict__ x1,
    const __bf16* __restrict__ WihP,  // packed fragments, KS = (D0+D1)/32
    const __bf16* __restrict__ WhhP,  // packed fragments, KS = 8
    const float* __restrict__ bih, const float* __restrict__ bhh,
    __bf16* __restrict__ out_bf, float* __restrict__ out_f32)
{
  constexpr int Dtot = D0 + D1;
  constexpr int KS0  = Dtot / 32;
  const int tid  = threadIdx.x;
  const int wave = tid >> 5;
  const int lane = tid & 31;
  const int l16  = lane & 15;
  const int hi   = (lane >= 16);

  // h := 0
  for (int i = tid; i < MT * H_SZ; i += 256) {
    int m = i >> 8, c = i & 255;
    sm.H[m][c] = 0.0f;
    sm.A[m][Dtot + c] = (__bf16)0.0f;
  }

  // per-lane tile metadata + biases (column = W row index in [0,768))
  int   col[NTPW];
  float bRZ[NTPW], bNI[NTPW], bNH[NTPW];
#pragma unroll
  for (int j = 0; j < NTPW; ++j) {
    int tIdx = wave * NTPW + j;
    int c = tIdx * 16 + l16;
    col[j] = c;
    if (tIdx < 32) { bRZ[j] = ld_gf(bih + c) + ld_gf(bhh + c); bNI[j] = 0.f; bNH[j] = 0.f; }
    else           { bRZ[j] = 0.f; bNI[j] = ld_gf(bih + c); bNH[j] = ld_gf(bhh + c); }
  }
  __syncthreads();

#pragma clang loop unroll(disable)
  for (int t = 0; t < T_SZ; ++t) {
    // Launder weight pointers: redefine per-iteration so the k-loop's global
    // loads are anchored inside the t-loop (no cross-iteration hoist/spill).
    const __bf16* WihPt = WihP;
    const __bf16* WhhPt = WhhP;
    asm volatile("" : "+s"(WihPt), "+s"(WhhPt));

    // ---- stage x_t tile (and concat source) into sm.A[:, 0:Dtot) ----
    {
#pragma unroll
      for (int i = tid * 4; i < MT * D0; i += 256 * 4) {
        int m = i / D0, k = i - m * D0;
        *(i32x2*)&sm.A[m][k] = ld_g64(&x0[((size_t)(b0 + m) * T_SZ + t) * D0 + k]);
      }
      if constexpr (D1 > 0) {
#pragma unroll
        for (int i = tid * 4; i < MT * D1; i += 256 * 4) {
          int m = i / D1, k = i - m * D1;
          *(i32x2*)&sm.A[m][D0 + k] = ld_g64(&x1[((size_t)(b0 + m) * T_SZ + t) * D1 + k]);
        }
      }
    }
    __syncthreads();

    v8f accA[NTPW], accN[NTPW];
    {
      v8f zz;
#pragma unroll
      for (int v = 0; v < 8; ++v) zz[v] = 0.f;
#pragma unroll
      for (int j = 0; j < NTPW; ++j) { accA[j] = zz; accN[j] = zz; }
    }

    // ---- input projection: K over Dtot (gi) ----
#pragma unroll
    for (int ks = 0; ks < KS0; ++ks) {
      int ko = ks * 32 + (hi ? 8 : 0);
      v16bf a;
      ((i32x4*)&a)[0] = *(const i32x4*)&sm.A[l16][ko];
      ((i32x4*)&a)[1] = *(const i32x4*)&sm.A[l16][ko + 16];
      v16bf bfr[NTPW];
#pragma unroll
      for (int j = 0; j < NTPW; ++j)
        bfr[j] = load_bfrag(WihPt, wave * NTPW + j, KS0, ks, lane);
#pragma unroll
      for (int j = 0; j < NTPW; ++j) accA[j] = wmma_bf16(a, bfr[j], accA[j]);
    }
    // ---- hidden projection: K over H=256 (gh) ----
#pragma unroll
    for (int ks = 0; ks < 8; ++ks) {
      int ko = Dtot + ks * 32 + (hi ? 8 : 0);
      v16bf a;
      ((i32x4*)&a)[0] = *(const i32x4*)&sm.A[l16][ko];
      ((i32x4*)&a)[1] = *(const i32x4*)&sm.A[l16][ko + 16];
      v16bf bfr[NTPW];
#pragma unroll
      for (int j = 0; j < NTPW; ++j)
        bfr[j] = load_bfrag(WhhPt, wave * NTPW + j, 8, ks, lane);
#pragma unroll
      for (int j = 0; j < NTPW; ++j) {
        if (wave * NTPW + j < 32) accA[j] = wmma_bf16(a, bfr[j], accA[j]);  // r,z
        else                      accN[j] = wmma_bf16(a, bfr[j], accN[j]);  // n: gh
      }
    }

    // ---- spill preacts (+bias) to LDS; C layout: lane->col, vgpr->row ----
    {
      int mb = hi ? 8 : 0;
#pragma unroll
      for (int j = 0; j < NTPW; ++j) {
        int tIdx = wave * NTPW + j;
        if (tIdx < 32) {
          int c = col[j];                       // 0..511 (r|z)
#pragma unroll
          for (int v = 0; v < 8; ++v) sm.P[mb + v][c] = accA[j][v] + bRZ[j];
        } else {
          int c = col[j] - 512;                 // 0..255 (n)
#pragma unroll
          for (int v = 0; v < 8; ++v) {
            sm.P[mb + v][512 + c] = accA[j][v] + bNI[j];
            sm.P[mb + v][768 + c] = accN[j][v] + bNH[j];
          }
        }
      }
    }
    __syncthreads();

    // ---- gate math, h update, write outputs ----
    {
      int c = tid;  // 0..255 == H
#pragma unroll 2
      for (int m = 0; m < MT; ++m) {
        float r  = 1.0f / (1.0f + __expf(-sm.P[m][c]));
        float z  = 1.0f / (1.0f + __expf(-sm.P[m][256 + c]));
        float n  = tanhf(sm.P[m][512 + c] + r * sm.P[m][768 + c]);
        float hp = sm.H[m][c];
        float hn = (1.0f - z) * n + z * hp;
        sm.H[m][c] = hn;
        __bf16 hb = (__bf16)hn;
        sm.A[m][Dtot + c] = hb;
        size_t o = ((size_t)(b0 + m) * T_SZ + t) * H_SZ + c;
        st_gbf(out_bf + o, hb);
        if constexpr (OUTF) st_gf(out_f32 + o, hn);
      }
    }
    __syncthreads();
  }
}

// crime + anomaly GRUs fused into one launch (independent -> double parallelism)
__global__ __launch_bounds__(256) void gru_dual_kernel(
    const __bf16* xc, const __bf16* WihC, const __bf16* WhhC,
    const float* bihC, const float* bhhC, __bf16* outC,
    const __bf16* xa, const __bf16* WihA, const __bf16* WhhA,
    const float* bihA, const float* bhhA, __bf16* outA, int nblkA)
{
  __shared__ SMem sm;
  if ((int)blockIdx.x < nblkA)
    gru_scan_body<128, 0, false>(sm, blockIdx.x * MT, xc, nullptr,
                                 WihC, WhhC, bihC, bhhC, outC, nullptr);
  else
    gru_scan_body<96, 0, false>(sm, (blockIdx.x - nblkA) * MT, xa, nullptr,
                                WihA, WhhA, bihA, bhhA, outA, nullptr);
}

__global__ __launch_bounds__(256) void gru_inter_kernel(
    const __bf16* outc, const __bf16* outa,
    const __bf16* WihI, const __bf16* WhhI,
    const float* bihI, const float* bhhI,
    __bf16* outi_bf, float* outi_f32)
{
  __shared__ SMem sm;
  gru_scan_body<256, 256, true>(sm, blockIdx.x * MT, outc, outa,
                                WihI, WhhI, bihI, bhhI, outi_bf, outi_f32);
}

// ---------------------------------------------------------------------------
// Attention scores via WMMA: um = tanh(out_i @ Wv^T + bv); score = um . wu
// N padded 40 -> 48 with zero rows; K = H = 256; M tiles of 16 B*T rows.
// ---------------------------------------------------------------------------
__global__ __launch_bounds__(256) void scores_kernel(
    const __bf16* __restrict__ outi_bf,  // [BT, 256]
    const __bf16* __restrict__ WvPk,     // packed fragments [3][8][32][16]
    const float* __restrict__ bv, const float* __restrict__ wu,
    float* __restrict__ scores, int BT)
{
  int wave = threadIdx.x >> 5, lane = threadIdx.x & 31;
  int l16 = lane & 15, hi = (lane >= 16);
  int mtile = blockIdx.x * 8 + wave;
  int row0 = mtile * 16;
  if (row0 >= BT) return;  // wave-uniform

  v8f acc[3] = {};
  float bvv[3], wuv[3];
#pragma unroll
  for (int tI = 0; tI < 3; ++tI) {
    int a = tI * 16 + l16;
    bvv[tI] = (a < 40) ? ld_gf(bv + a) : 0.f;
    wuv[tI] = (a < 40) ? ld_gf(wu + a) : 0.f;
  }
#pragma unroll
  for (int ks = 0; ks < 8; ++ks) {
    const __bf16* pa = outi_bf + (size_t)(row0 + l16) * 256 + ks * 32 + (hi ? 8 : 0);
    v16bf a;
    ((i32x4*)&a)[0] = ld_g128(pa);
    ((i32x4*)&a)[1] = ld_g128(pa + 16);
    v16bf bfr[3];
#pragma unroll
    for (int tI = 0; tI < 3; ++tI) bfr[tI] = load_bfrag(WvPk, tI, 8, ks, lane);
#pragma unroll
    for (int tI = 0; tI < 3; ++tI) acc[tI] = wmma_bf16(a, bfr[tI], acc[tI]);
  }
  // per-lane partial: sum over its 3 columns of tanh(um)*wu, then 16-lane reduce
  float s[8];
#pragma unroll
  for (int v = 0; v < 8; ++v) {
    float p = 0.f;
#pragma unroll
    for (int tI = 0; tI < 3; ++tI) p += tanhf(acc[tI][v] + bvv[tI]) * wuv[tI];
    p += __shfl_xor(p, 1); p += __shfl_xor(p, 2);
    p += __shfl_xor(p, 4); p += __shfl_xor(p, 8);
    s[v] = p;
  }
  int mb = hi ? 8 : 0;
  if (l16 == 0) {
#pragma unroll
    for (int v = 0; v < 8; ++v) st_gf(scores + row0 + mb + v, s[v]);
  }
}

// softmax over T + context = sum_t alpha_t * out_i[:,t,:]
__global__ __launch_bounds__(128) void attn_kernel(
    const float* __restrict__ scores,  // [B, T]
    const float* __restrict__ outi,    // [B, T, 256] f32
    float* __restrict__ ctx)           // [B, 256]
{
  __shared__ float stmp[128];
  __shared__ float sAl[T_SZ];
  __shared__ float red[2];
  int b = blockIdx.x, tid = threadIdx.x;
  float sv = (tid < T_SZ) ? ld_gf(scores + b * T_SZ + tid) : -1e30f;
  stmp[tid] = sv;
  __syncthreads();
  if (tid == 0) {
    float mx = -1e30f;
    for (int i = 0; i < T_SZ; ++i) mx = fmaxf(mx, stmp[i]);
    red[0] = mx;
  }
  __syncthreads();
  float e = (tid < T_SZ) ? __expf(sv - red[0]) : 0.f;
  stmp[tid] = e;
  __syncthreads();
  if (tid == 0) {
    float sm = 0.f;
    for (int i = 0; i < T_SZ; ++i) sm += stmp[i];
    red[1] = sm;
  }
  __syncthreads();
  if (tid < T_SZ) sAl[tid] = e / red[1];
  __syncthreads();
#pragma unroll 2
  for (int h = tid; h < H_SZ; h += 128) {
    float acc = 0.f;
    for (int t = 0; t < T_SZ; ++t)
      acc += sAl[t] * ld_gf(&outi[((size_t)b * T_SZ + t) * H_SZ + h]);
    st_gf(ctx + (size_t)b * H_SZ + h, acc);
  }
}

// ---------------------------------------------------------------------------
__global__ void cvt_bf16(const float* __restrict__ src, __bf16* __restrict__ dst,
                         size_t n) {
  size_t i = (size_t)blockIdx.x * blockDim.x + threadIdx.x;
  size_t st = (size_t)gridDim.x * blockDim.x;
  for (; i < n; i += st) st_gbf(dst + i, (__bf16)ld_gf(src + i));
}

// repack W [rows, K] f32 row-major -> bf16 WMMA B-fragment-major layout
// dst[((tile*KS + ks)*32 + ((k>>3)&1)*16 + (col&15)) * 16 + ((k&7) | (((k>>4)&1)<<3))]
__global__ void pack_w(const float* __restrict__ W, __bf16* __restrict__ Wpk,
                       int K, int rows, int srcRows) {
  size_t n = (size_t)rows * K;
  int KS = K >> 5;
  size_t i = (size_t)blockIdx.x * blockDim.x + threadIdx.x;
  size_t st = (size_t)gridDim.x * blockDim.x;
  for (; i < n; i += st) {
    int c = (int)(i / K);        // W row (output column)
    int k = (int)(i - (size_t)c * K);
    float v = (c < srcRows) ? ld_gf(W + (size_t)c * K + k) : 0.f;
    int tile = c >> 4, l16 = c & 15;
    int ks = k >> 5, kin = k & 31, hib = (kin >> 3) & 1, e = (kin & 7) | ((kin >> 4) << 3);
    size_t dst = ((((size_t)tile * KS + ks) * 32) + hib * 16 + l16) * 16 + e;
    st_gbf(Wpk + dst, (__bf16)v);
  }
}

// ---------------------------------------------------------------------------
extern "C" void kernel_launch(void* const* d_in, const int* in_sizes, int n_in,
                              void* d_out, int out_size, void* d_ws, size_t ws_size,
                              hipStream_t stream)
{
  const float* x_c   = (const float*)d_in[0];
  const float* x_a   = (const float*)d_in[1];
  const float* Wih_c = (const float*)d_in[2];
  const float* Whh_c = (const float*)d_in[3];
  const float* bih_c = (const float*)d_in[4];
  const float* bhh_c = (const float*)d_in[5];
  const float* Wih_a = (const float*)d_in[6];
  const float* Whh_a = (const float*)d_in[7];
  const float* bih_a = (const float*)d_in[8];
  const float* bhh_a = (const float*)d_in[9];
  const float* Wih_i = (const float*)d_in[10];
  const float* Whh_i = (const float*)d_in[11];
  const float* bih_i = (const float*)d_in[12];
  const float* bhh_i = (const float*)d_in[13];
  const float* Wv    = (const float*)d_in[14];
  const float* bv    = (const float*)d_in[15];
  const float* wu    = (const float*)d_in[16];
  float* ctx = (float*)d_out;

  const size_t BT = (size_t)B_SZ * T_SZ;
  uint8_t* ws = (uint8_t*)d_ws;
  size_t off = 0;
  auto take = [&](size_t elems, size_t esz) -> void* {
    void* p = ws + off;
    off = (off + elems * esz + 255) & ~(size_t)255;
    return p;
  };
  __bf16* xc_bf   = (__bf16*)take(BT * 128, 2);
  __bf16* xa_bf   = (__bf16*)take(BT * 96, 2);
  __bf16* wihc_pk = (__bf16*)take((size_t)G3 * 128, 2);
  __bf16* whhc_pk = (__bf16*)take((size_t)G3 * 256, 2);
  __bf16* wiha_pk = (__bf16*)take((size_t)G3 * 96, 2);
  __bf16* whha_pk = (__bf16*)take((size_t)G3 * 256, 2);
  __bf16* wihi_pk = (__bf16*)take((size_t)G3 * 512, 2);
  __bf16* whhi_pk = (__bf16*)take((size_t)G3 * 256, 2);
  __bf16* wvp_pk  = (__bf16*)take(48 * 256, 2);
  __bf16* outc_bf = (__bf16*)take(BT * 256, 2);
  __bf16* outa_bf = (__bf16*)take(BT * 256, 2);
  __bf16* outi_bf = (__bf16*)take(BT * 256, 2);
  float*  outi_f  = (float*) take(BT * 256, 4);
  float*  scoresb = (float*) take(BT, 4);

  cvt_bf16<<<4096, 256, 0, stream>>>(x_c, xc_bf, BT * 128);
  cvt_bf16<<<4096, 256, 0, stream>>>(x_a, xa_bf, BT * 96);
  pack_w<<<384, 256, 0, stream>>>(Wih_c, wihc_pk, 128, G3, G3);
  pack_w<<<384, 256, 0, stream>>>(Whh_c, whhc_pk, 256, G3, G3);
  pack_w<<<384, 256, 0, stream>>>(Wih_a, wiha_pk, 96, G3, G3);
  pack_w<<<384, 256, 0, stream>>>(Whh_a, whha_pk, 256, G3, G3);
  pack_w<<<384, 256, 0, stream>>>(Wih_i, wihi_pk, 512, G3, G3);
  pack_w<<<384, 256, 0, stream>>>(Whh_i, whhi_pk, 256, G3, G3);
  pack_w<<<48, 256, 0, stream>>>(Wv, wvp_pk, 256, 48, 40);

  // stage 1: crime + anomaly GRU scans (fused, 256 blocks)
  gru_dual_kernel<<<256, 256, 0, stream>>>(
      xc_bf, wihc_pk, whhc_pk, bih_c, bhh_c, outc_bf,
      xa_bf, wiha_pk, whha_pk, bih_a, bhh_a, outa_bf, 128);
  // stage 2: inter GRU over concat(out_c, out_a)
  gru_inter_kernel<<<128, 256, 0, stream>>>(
      outc_bf, outa_bf, wihi_pk, whhi_pk, bih_i, bhh_i, outi_bf, outi_f);
  // stage 3: attention scores (WMMA)
  scores_kernel<<<1920, 256, 0, stream>>>(outi_bf, wvp_pk, bv, wu, scoresb, (int)BT);
  // stage 4: softmax + context
  attn_kernel<<<2048, 128, 0, stream>>>(scoresb, outi_f, ctx);
}
